// EGNNStateModel_48172353192161
// MI455X (gfx1250) — compile-verified, matrix operand order checked by reading, not run on hardware
//
#include <hip/hip_runtime.h>
#include <hip/hip_bf16.h>
#include <math.h>

typedef __bf16 bf16;
typedef __attribute__((ext_vector_type(16))) __bf16 v16bf;
typedef __attribute__((ext_vector_type(8)))  float  v8f;

#define N_NODES 50000
#define N_EDGES 800000
#define NLAYERS 6
#define F_DIM 32
#define M_DIM 64
#define EIN   69      // 2F + EA + 1
#define H1    138     // 2*EIN
#define KP1   96      // EIN padded to 3*32
#define NP1   160     // H1 padded to 10*16 (and 5*32 for GEMM2 K)
#define RHID  128

// ---------------- WMMA fragment helpers (wave32, 16x16x32 bf16) ----------------
// A/B 16-bit operand lane layout (ISA 7.12.2): lane<16 holds K = {0..7,16..23},
// lane>=16 holds K = {8..15,24..31} for each 32-deep k-block.
__device__ __forceinline__ int kidx(int i, int hi) {
    return (i < 8 ? i : i + 8) + (hi ? 8 : 0);
}

__device__ __forceinline__ v8f zero8() {
    v8f r;
#pragma unroll
    for (int i = 0; i < 8; ++i) r[i] = 0.f;
    return r;
}

__device__ __forceinline__ v16bf load_a_frag(const bf16* row, int kb, int hi) {
    v16bf a;
#pragma unroll
    for (int i = 0; i < 16; ++i) a[i] = row[kb * 32 + kidx(i, hi)];
    return a;
}

// Weights are pre-packed so a lane's 16 halves are contiguous (32B) per chunk.
__device__ __forceinline__ v16bf load_b_frag(const bf16* frag, int chunk, int lane) {
    return *(const v16bf*)(frag + (size_t)chunk * 512 + lane * 16);
}

__device__ __forceinline__ float silu_f(float x) { return x / (1.f + __expf(-x)); }
__device__ __forceinline__ float gelu_f(float x) {
    return 0.5f * x * (1.f + erff(x * 0.70710678118654752f));
}

// ---------------- prep kernels ----------------
__global__ void pack_weight_kernel(const float* __restrict__ src, bf16* __restrict__ dst,
                                   int Ks, int Ns, int Kp, int Np) {
    int tid = blockIdx.x * blockDim.x + threadIdx.x;
    int ncn = Np >> 4;
    int total = (Kp >> 5) * ncn * 512;
    if (tid >= total) return;
    int chunk = tid >> 9;
    int within = tid & 511;
    int lane = within >> 4;
    int elem = within & 15;
    int kb = chunk / ncn, nb = chunk % ncn;
    int k = kb * 32 + kidx(elem, lane >= 16);
    int n = nb * 16 + (lane & 15);
    float v = (k < Ks && n < Ns) ? src[(size_t)k * Ns + n] : 0.f;
    dst[tid] = (bf16)v;
}

__global__ void prep_eattr_kernel(const float* __restrict__ pos, const int* __restrict__ eidx,
                                  const float* __restrict__ ea, bf16* __restrict__ out) {
    int e = blockIdx.x * blockDim.x + threadIdx.x;
    if (e >= N_EDGES) return;
    int s = eidx[e];              // src (x_j)
    int d = eidx[N_EDGES + e];    // dst (x_i)
    float rx = pos[(size_t)s * 2 + 0] - pos[(size_t)d * 2 + 0];
    float ry = pos[(size_t)s * 2 + 1] - pos[(size_t)d * 2 + 1];
    float rd = rx * rx + ry * ry;
#pragma unroll
    for (int j = 0; j < 4; ++j) out[(size_t)e * 5 + j] = (bf16)ea[(size_t)e * 4 + j];
    out[(size_t)e * 5 + 4] = (bf16)rd;
}

__global__ void prep_feats_kernel(const float* __restrict__ x, float* __restrict__ feats,
                                  bf16* __restrict__ fbf) {
    int i = blockIdx.x * blockDim.x + threadIdx.x;
    if (i >= N_NODES * F_DIM) return;
    float v = x[i];
    feats[i] = v;
    fbf[i] = (bf16)v;
}

__global__ void zero_f32_kernel(float* __restrict__ p, int n) {
    int i = blockIdx.x * blockDim.x + threadIdx.x;
    if (i < n) p[i] = 0.f;
}

__global__ void lnstats_kernel(const float* __restrict__ feats, float* __restrict__ stats, int n) {
    __shared__ float s1[256], s2[256];
    float a = 0.f, b = 0.f;
    for (int i = blockIdx.x * blockDim.x + threadIdx.x; i < n; i += gridDim.x * blockDim.x) {
        float v = feats[i];
        a += v;
        b += v * v;
    }
    s1[threadIdx.x] = a; s2[threadIdx.x] = b;
    __syncthreads();
    for (int off = 128; off > 0; off >>= 1) {
        if ((int)threadIdx.x < off) {
            s1[threadIdx.x] += s1[threadIdx.x + off];
            s2[threadIdx.x] += s2[threadIdx.x + off];
        }
        __syncthreads();
    }
    if (threadIdx.x == 0) {
        atomicAdd(&stats[0], s1[0]);
        atomicAdd(&stats[1], s2[0]);
    }
}

// ---------------- edge MLP + scatter-add (4 waves/block, 16 edges/wave) ----------------
__global__ void __launch_bounds__(128)
edge_kernel(const bf16* __restrict__ fbf, const bf16* __restrict__ eattr,
            const int* __restrict__ eidx, float* __restrict__ m_i,
            const bf16* __restrict__ w1f, const bf16* __restrict__ w2f,
            const float* __restrict__ b1, const float* __restrict__ b2) {
    __shared__ __align__(16) bf16 Abuf[4][16][KP1];
    __shared__ __align__(16) bf16 Hbuf[4][16][NP1];
    __shared__ int dnode[4][16];

    int lane = threadIdx.x & 31, wave = threadIdx.x >> 5;
    int hi = lane >> 4;
    int r = lane & 15;
    int tileBase = (blockIdx.x * 4 + wave) * 16;
    int e = tileBase + r;
    bool valid = e < N_EDGES;

    if (hi == 0) {
        int d = valid ? eidx[N_EDGES + e] : -1;
        dnode[wave][r] = d;
        uint4* dp = (uint4*)&Abuf[wave][r][0];
        if (valid) {
            const uint4* sp = (const uint4*)(fbf + (size_t)d * F_DIM);
#pragma unroll
            for (int i = 0; i < 4; ++i) dp[i] = sp[i];
        } else {
            uint4 z = {0u, 0u, 0u, 0u};
#pragma unroll
            for (int i = 0; i < 4; ++i) dp[i] = z;
        }
    } else {
        int s = valid ? eidx[e] : -1;
        uint4* dp = (uint4*)&Abuf[wave][r][F_DIM];
        if (valid) {
            const uint4* sp = (const uint4*)(fbf + (size_t)s * F_DIM);
#pragma unroll
            for (int i = 0; i < 4; ++i) dp[i] = sp[i];
        } else {
            uint4 z = {0u, 0u, 0u, 0u};
#pragma unroll
            for (int i = 0; i < 4; ++i) dp[i] = z;
        }
#pragma unroll
        for (int j = 0; j < 5; ++j)
            Abuf[wave][r][64 + j] = valid ? eattr[(size_t)e * 5 + j] : (bf16)0.f;
#pragma unroll
        for (int j = 5; j < 32; ++j) Abuf[wave][r][64 + j] = (bf16)0.f;
    }
    __syncthreads();

    // GEMM1: [16x96] x [96x160], zero-init acc (inline C=0), bias+silu in epilogue
    const bf16* arow = &Abuf[wave][lane & 15][0];
#pragma unroll
    for (int nt = 0; nt < 10; ++nt) {
        int col = nt * 16 + (lane & 15);
        float bv = (col < H1) ? b1[col] : 0.f;
        v8f acc = zero8();
#pragma unroll
        for (int kb = 0; kb < 3; ++kb) {
            v16bf a = load_a_frag(arow, kb, hi);
            v16bf b = load_b_frag(w1f, kb * 10 + nt, lane);
            acc = __builtin_amdgcn_wmma_f32_16x16x32_bf16(false, a, false, b, (short)0, acc,
                                                          false, false);
        }
#pragma unroll
        for (int i = 0; i < 8; ++i)
            Hbuf[wave][i + 8 * hi][col] = (bf16)silu_f(acc[i] + bv);
    }
    __syncthreads();

    // GEMM2: [16x160] x [160x64], bias+silu in epilogue, atomic scatter-add into m_i[dst]
    const bf16* hrow = &Hbuf[wave][lane & 15][0];
#pragma unroll
    for (int nt = 0; nt < 4; ++nt) {
        int col = nt * 16 + (lane & 15);
        float bv = b2[col];
        v8f acc = zero8();
#pragma unroll
        for (int kb = 0; kb < 5; ++kb) {
            v16bf a = load_a_frag(hrow, kb, hi);
            v16bf b = load_b_frag(w2f, kb * 4 + nt, lane);
            acc = __builtin_amdgcn_wmma_f32_16x16x32_bf16(false, a, false, b, (short)0, acc,
                                                          false, false);
        }
#pragma unroll
        for (int i = 0; i < 8; ++i) {
            int m = i + 8 * hi;
            int node = dnode[wave][m];
            if (node >= 0)
                atomicAdd(&m_i[(size_t)node * M_DIM + col], silu_f(acc[i] + bv));
        }
    }
}

// ---------------- node update: graph-LN + MLP + residual ----------------
__global__ void __launch_bounds__(128)
node_kernel(float* __restrict__ feats, bf16* __restrict__ fbf,
            const float* __restrict__ m_i, const float* __restrict__ stats,
            const float* __restrict__ g, const float* __restrict__ bta,
            const bf16* __restrict__ w1f, const bf16* __restrict__ w2f,
            const float* __restrict__ b1, const float* __restrict__ b2) {
    __shared__ __align__(16) bf16 Abuf[4][16][96];
    __shared__ __align__(16) bf16 Hbuf[4][16][64];
    int lane = threadIdx.x & 31, wave = threadIdx.x >> 5;
    int hi = lane >> 4, r = lane & 15;
    int tileBase = (blockIdx.x * 4 + wave) * 16;

    float cnt = (float)N_NODES * (float)F_DIM;
    float mean = stats[0] / cnt;
    float var = stats[1] / cnt - mean * mean;
    float rinv = rsqrtf(var + 1e-5f);

    int node = tileBase + r;
    bool valid = node < N_NODES;
    if (hi == 0) {
        if (valid) {
            const float4* sp = (const float4*)(feats + (size_t)node * F_DIM);
#pragma unroll
            for (int q = 0; q < 8; ++q) {
                float4 v4 = sp[q];
                float vv[4] = {v4.x, v4.y, v4.z, v4.w};
#pragma unroll
                for (int j = 0; j < 4; ++j) {
                    int f = q * 4 + j;
                    float v = (vv[j] - mean) * rinv;
                    Abuf[wave][r][f] = (bf16)(g[f] * v + bta[f]);
                }
            }
        } else {
#pragma unroll
            for (int f = 0; f < F_DIM; ++f) Abuf[wave][r][f] = (bf16)0.f;
        }
    } else {
        if (valid) {
            const float4* sp = (const float4*)(m_i + (size_t)node * M_DIM);
#pragma unroll
            for (int q = 0; q < 16; ++q) {
                float4 v4 = sp[q];
                Abuf[wave][r][F_DIM + q * 4 + 0] = (bf16)v4.x;
                Abuf[wave][r][F_DIM + q * 4 + 1] = (bf16)v4.y;
                Abuf[wave][r][F_DIM + q * 4 + 2] = (bf16)v4.z;
                Abuf[wave][r][F_DIM + q * 4 + 3] = (bf16)v4.w;
            }
        } else {
#pragma unroll
            for (int j = 0; j < M_DIM; ++j) Abuf[wave][r][F_DIM + j] = (bf16)0.f;
        }
    }
    __syncthreads();

    // GEMM1: [16x96] x [96x64], bias+silu in epilogue
    const bf16* arow = &Abuf[wave][lane & 15][0];
#pragma unroll
    for (int nt = 0; nt < 4; ++nt) {
        int col = nt * 16 + (lane & 15);
        float bv = b1[col];
        v8f acc = zero8();
#pragma unroll
        for (int kb = 0; kb < 3; ++kb) {
            v16bf a = load_a_frag(arow, kb, hi);
            v16bf b = load_b_frag(w1f, kb * 4 + nt, lane);
            acc = __builtin_amdgcn_wmma_f32_16x16x32_bf16(false, a, false, b, (short)0, acc,
                                                          false, false);
        }
#pragma unroll
        for (int i = 0; i < 8; ++i)
            Hbuf[wave][i + 8 * hi][col] = (bf16)silu_f(acc[i] + bv);
    }
    __syncthreads();

    // GEMM2: [16x64] x [64x32], bias in epilogue, residual
    const bf16* hrow = &Hbuf[wave][lane & 15][0];
#pragma unroll
    for (int nt = 0; nt < 2; ++nt) {
        int col = nt * 16 + (lane & 15);
        float bv = b2[col];
        v8f acc = zero8();
#pragma unroll
        for (int kb = 0; kb < 2; ++kb) {
            v16bf a = load_a_frag(hrow, kb, hi);
            v16bf b = load_b_frag(w2f, kb * 2 + nt, lane);
            acc = __builtin_amdgcn_wmma_f32_16x16x32_bf16(false, a, false, b, (short)0, acc,
                                                          false, false);
        }
#pragma unroll
        for (int i = 0; i < 8; ++i) {
            int m = i + 8 * hi;
            int nd = tileBase + m;
            if (nd < N_NODES) {
                size_t idx = (size_t)nd * F_DIM + col;
                float nf = feats[idx] + acc[i] + bv;
                feats[idx] = nf;
                fbf[idx] = (bf16)nf;
            }
        }
    }
}

// ---------------- readout: gelu MLP -> 5 outputs + feats copy ----------------
__global__ void __launch_bounds__(128)
readout_kernel(const float* __restrict__ feats, const bf16* __restrict__ fbf,
               const bf16* __restrict__ w1f, const bf16* __restrict__ w2f,
               const float* __restrict__ b1, const float* __restrict__ b2,
               float* __restrict__ out) {
    __shared__ __align__(16) bf16 Abuf[4][16][32];
    __shared__ __align__(16) bf16 Hbuf[4][16][RHID];
    int lane = threadIdx.x & 31, wave = threadIdx.x >> 5;
    int hi = lane >> 4, r = lane & 15;
    int tileBase = (blockIdx.x * 4 + wave) * 16;
    int node = tileBase + r;
    bool valid = node < N_NODES;

    if (hi == 0) {
        uint4* dp = (uint4*)&Abuf[wave][r][0];
        if (valid) {
            const uint4* sp = (const uint4*)(fbf + (size_t)node * F_DIM);
#pragma unroll
            for (int i = 0; i < 4; ++i) dp[i] = sp[i];
        } else {
            uint4 z = {0u, 0u, 0u, 0u};
#pragma unroll
            for (int i = 0; i < 4; ++i) dp[i] = z;
        }
    } else if (valid) {
        // feats (5th output) straight copy, f32
        const float4* sp = (const float4*)(feats + (size_t)node * F_DIM);
        float4* dp = (float4*)(out + (size_t)5 * N_NODES + (size_t)node * F_DIM);
#pragma unroll
        for (int i = 0; i < 8; ++i) dp[i] = sp[i];
    }
    __syncthreads();

    // GEMM1: [16x32] x [32x128], bias + exact gelu in epilogue
    const bf16* arow = &Abuf[wave][lane & 15][0];
#pragma unroll
    for (int nt = 0; nt < 8; ++nt) {
        int col = nt * 16 + (lane & 15);
        float bv = b1[col];
        v8f acc = zero8();
        v16bf a = load_a_frag(arow, 0, hi);
        v16bf b = load_b_frag(w1f, nt, lane);
        acc = __builtin_amdgcn_wmma_f32_16x16x32_bf16(false, a, false, b, (short)0, acc,
                                                      false, false);
#pragma unroll
        for (int i = 0; i < 8; ++i)
            Hbuf[wave][i + 8 * hi][col] = (bf16)gelu_f(acc[i] + bv);
    }
    __syncthreads();

    // GEMM2: [16x128] x [128x16] (cols >= 5 padded with zero weights)
    const bf16* hrow = &Hbuf[wave][lane & 15][0];
    int col = lane & 15;
    float bv = (col < 5) ? b2[col] : 0.f;
    v8f acc = zero8();
#pragma unroll
    for (int kb = 0; kb < 4; ++kb) {
        v16bf a = load_a_frag(hrow, kb, hi);
        v16bf b = load_b_frag(w2f, kb, lane);
        acc = __builtin_amdgcn_wmma_f32_16x16x32_bf16(false, a, false, b, (short)0, acc,
                                                      false, false);
    }
#pragma unroll
    for (int i = 0; i < 8; ++i) {
        int nd = tileBase + i + 8 * hi;
        if (nd < N_NODES && col < 5) {
            float v = acc[i] + bv;
            if (col == 0)      out[nd] = v;
            else if (col == 1) out[N_NODES + nd] = v;
            else if (col == 2) out[2 * N_NODES + nd] = v;
            else if (col == 3) out[3 * N_NODES + (size_t)nd * 2 + 0] = v;
            else               out[3 * N_NODES + (size_t)nd * 2 + 1] = v;
        }
    }
}

// ---------------- host launcher ----------------
extern "C" void kernel_launch(void* const* d_in, const int* in_sizes, int n_in,
                              void* d_out, int out_size, void* d_ws, size_t ws_size,
                              hipStream_t stream) {
    (void)in_sizes; (void)n_in; (void)out_size; (void)ws_size;
    const float* x    = (const float*)d_in[0];
    const float* pos  = (const float*)d_in[1];
    const int*   eidx = (const int*)d_in[2];
    const float* ea   = (const float*)d_in[3];
    const float* ew1  = (const float*)d_in[5];
    const float* eb1  = (const float*)d_in[6];
    const float* ew2  = (const float*)d_in[7];
    const float* eb2  = (const float*)d_in[8];
    const float* ln_g = (const float*)d_in[9];
    const float* ln_b = (const float*)d_in[10];
    const float* nw1  = (const float*)d_in[11];
    const float* nb1  = (const float*)d_in[12];
    const float* nw2  = (const float*)d_in[13];
    const float* nb2  = (const float*)d_in[14];
    const float* rw1  = (const float*)d_in[15];
    const float* rb1  = (const float*)d_in[16];
    const float* rw2  = (const float*)d_in[17];
    const float* rb2  = (const float*)d_in[18];

    char* ws = (char*)d_ws;
    size_t off = 0;
    auto take = [&](size_t bytes) { size_t o = off; off += (bytes + 255) & ~(size_t)255; return o; };
    float* feats = (float*)(ws + take((size_t)N_NODES * F_DIM * 4));
    bf16*  fbf   = (bf16*) (ws + take((size_t)N_NODES * F_DIM * 2));
    float* m_i   = (float*)(ws + take((size_t)N_NODES * M_DIM * 4));
    bf16*  eatb  = (bf16*) (ws + take((size_t)N_EDGES * 5 * 2));
    float* stats = (float*)(ws + take(256));
    bf16*  wf    = (bf16*) (ws + take((size_t)208896 * 2));

    // packed-weight layout (in bf16 halves)
    const size_t W1F = 15360, W2F = 10240, NW1F = 6144, NW2F = 2048;
    const size_t LSTRIDE = W1F + W2F + NW1F + NW2F; // 33792
    bf16* rw1f = wf + NLAYERS * LSTRIDE;            // 202752
    bf16* rw2f = rw1f + 4096;                       // 206848

    auto pack = [&](const float* src, bf16* dst, int Ks, int Ns, int Kp, int Np) {
        int total = (Kp / 32) * (Np / 16) * 512;
        pack_weight_kernel<<<(total + 255) / 256, 256, 0, stream>>>(src, dst, Ks, Ns, Kp, Np);
    };
    for (int l = 0; l < NLAYERS; ++l) {
        bf16* base = wf + l * LSTRIDE;
        pack(ew1 + (size_t)l * EIN * H1,          base,                EIN, H1,    KP1, NP1);
        pack(ew2 + (size_t)l * H1 * M_DIM,        base + W1F,          H1,  M_DIM, NP1, 64);
        pack(nw1 + (size_t)l * 96 * 64,           base + W1F + W2F,    96,  64,    96,  64);
        pack(nw2 + (size_t)l * 64 * 32,           base + W1F + W2F + NW1F, 64, 32, 64,  32);
    }
    pack(rw1, rw1f, 32, RHID, 32, RHID);
    pack(rw2, rw2f, RHID, 5, RHID, 16);

    prep_eattr_kernel<<<(N_EDGES + 255) / 256, 256, 0, stream>>>(pos, eidx, ea, eatb);
    prep_feats_kernel<<<(N_NODES * F_DIM + 255) / 256, 256, 0, stream>>>(x, feats, fbf);

    const int edgeBlocks = (N_EDGES + 63) / 64;
    const int nodeBlocks = (N_NODES + 63) / 64;
    for (int l = 0; l < NLAYERS; ++l) {
        bf16* base = wf + l * LSTRIDE;
        zero_f32_kernel<<<(N_NODES * M_DIM + 255) / 256, 256, 0, stream>>>(m_i, N_NODES * M_DIM);
        zero_f32_kernel<<<1, 32, 0, stream>>>(stats, 2);
        edge_kernel<<<edgeBlocks, 128, 0, stream>>>(fbf, eatb, eidx, m_i,
                                                    base, base + W1F,
                                                    eb1 + (size_t)l * H1, eb2 + (size_t)l * M_DIM);
        lnstats_kernel<<<256, 256, 0, stream>>>(feats, stats, N_NODES * F_DIM);
        node_kernel<<<nodeBlocks, 128, 0, stream>>>(feats, fbf, m_i, stats,
                                                    ln_g + (size_t)l * F_DIM, ln_b + (size_t)l * F_DIM,
                                                    base + W1F + W2F, base + W1F + W2F + NW1F,
                                                    nb1 + (size_t)l * 64, nb2 + (size_t)l * 32);
    }
    readout_kernel<<<nodeBlocks, 128, 0, stream>>>(feats, fbf, rw1f, rw2f, rb1, rb2, (float*)d_out);
}
